// MultiHeadAttention_84842783965536
// MI455X (gfx1250) — matrix operand order repaired and checked by reading, not compiled, not run on hardware
//
#include <hip/hip_runtime.h>
#include <hip/hip_bf16.h>

typedef __attribute__((ext_vector_type(16))) __bf16 v16bf;
typedef __attribute__((ext_vector_type(8)))  float  v8f;
typedef __attribute__((ext_vector_type(4)))  unsigned int u32x4;
typedef __attribute__((ext_vector_type(8)))  int          i32x8;
typedef __attribute__((ext_vector_type(4)))  int          i32x4;

union Frag16 {
    v16bf v;
    unsigned int u[8];
};

union Acc {
    v8f v;
    float f[8];
};

static __device__ __forceinline__ unsigned short f32_to_bf16(float x) {
    unsigned int u = __builtin_bit_cast(unsigned int, x);
    u += 0x7FFFu + ((u >> 16) & 1u);            // round-to-nearest-even
    return (unsigned short)(u >> 16);
}

static __device__ __forceinline__ unsigned int pack_bf16x2(float lo, float hi) {
    return (unsigned int)f32_to_bf16(lo) | ((unsigned int)f32_to_bf16(hi) << 16);
}

// A-matrix (16x32 bf16) per-lane K offset, ISA 7.12.2:
// lanes 0-15: V0..3 hold K=0..7, V4..7 hold K=16..23; lanes 16-31: +8
static __device__ __forceinline__ int a_k0(int v, int g) {
    return (v < 4) ? (2 * v + 8 * g) : (16 + 2 * (v - 4) + 8 * g);
}

static __device__ __forceinline__ Frag16 load_a_frag(const unsigned short* p, int ldk, int lane) {
    Frag16 f;
    const int m = lane & 15, g = lane >> 4;
#pragma unroll
    for (int v = 0; v < 8; ++v)
        f.u[v] = *(const unsigned int*)(p + m * ldk + a_k0(v, g));
    return f;
}

// B fragment (32x16) from bf16 stored [n][k], row stride ldk.
// Lane l = column n=l%16; lanes 0-15: V0..3 K=0..7, V4..7 K=16..23; lanes 16-31: +8 (2/VGPR).
static __device__ __forceinline__ Frag16 load_b_frag(const unsigned short* p, int ldk, int lane) {
    Frag16 f;
    const int n = lane & 15, g = lane >> 4;
#pragma unroll
    for (int v = 0; v < 8; ++v)
        f.u[v] = *(const unsigned int*)(p + n * ldk + a_k0(v, g));
    return f;
}

static __device__ __forceinline__ v8f wmma_bf16(const Frag16& a, const Frag16& b, v8f c) {
    return __builtin_amdgcn_wmma_f32_16x16x32_bf16(
        false, a.v, false, b.v, (short)0, c, false, false);
}

static __device__ __forceinline__ float half_reduce_max(float x) {
#pragma unroll
    for (int m = 1; m <= 8; m <<= 1)
        x = fmaxf(x, __shfl_xor(x, m, 32));
    return x;
}

static __device__ __forceinline__ float half_reduce_sum(float x) {
#pragma unroll
    for (int m = 1; m <= 8; m <<= 1)
        x += __shfl_xor(x, m, 32);
    return x;
}

// ---------------------------------------------------------------------------
// Tensor Data Mover: 2D bf16 tile (tile_d0 elems/row, tile_d1 rows, row stride
// stride_el elements) from global -> LDS.  D# built per ISA ch.8 (group0/1),
// groups 2/3 zero (<=2D tensor).  6-arg builtin (clang-23 form).
// ---------------------------------------------------------------------------
static __device__ __forceinline__ void tdm_load_2d_bf16(
    unsigned int lds_byte_addr, const void* gptr,
    unsigned int tile_d0, unsigned int tile_d1,
    unsigned int tensor_d0, unsigned int tensor_d1,
    unsigned long long stride_el) {
    const unsigned long long ga = (unsigned long long)(size_t)gptr;
    u32x4 g0;
    g0[0] = 1u;                                            // count=1, user mode
    g0[1] = lds_byte_addr;                                 // lds_addr
    g0[2] = (unsigned int)(ga & 0xFFFFFFFFu);              // global_addr[31:0]
    g0[3] = (unsigned int)((ga >> 32) & 0x1FFFFFFu)        // global_addr[56:32]
          | (2u << 30);                                    // type=2 ("image")
    i32x8 g1;
    g1[0] = (int)(1u << 16);                               // data_size=1 (2B), wg_mask=0
    g1[1] = (int)((tensor_d0 & 0xFFFFu) << 16);            // tensor_dim0[15:0] @bit48
    g1[2] = (int)((tensor_d0 >> 16) | ((tensor_d1 & 0xFFFFu) << 16));
    g1[3] = (int)((tensor_d1 >> 16) | (tile_d0 << 16));    // tile_dim0 @bit112
    g1[4] = (int)(tile_d1 & 0xFFFFu);                      // tile_dim1 @bit128, tile_dim2=0
    g1[5] = (int)(stride_el & 0xFFFFFFFFull);              // tensor_dim0_stride[31:0]
    g1[6] = (int)((stride_el >> 32) & 0xFFFFull);          // stride[47:32], dim1_stride=0
    g1[7] = 0;
    const i32x4 z4 = {0, 0, 0, 0};
    const i32x8 z8 = {0, 0, 0, 0, 0, 0, 0, 0};
    __builtin_amdgcn_tensor_load_to_lds(g0, g1, z4, z4, z8, 0);
}

// ---------------------------------------------------------------------------
// GEMM:  Out[M,N] = A[M,K] * W[N,K]^T   (einsum 'mk,nk->mn')
// Block tile 128x128, 8 waves = 4(M) x 2(N), wave tile 32x64, K-step 32.
// Double-buffered LDS.  A_BF16: A staged via TDM; else fp32 staged via regs
// with convert.  W (fp32) always staged via regs.  OUT_BF16 or fp32+bias.
// ---------------------------------------------------------------------------
template <bool A_BF16, bool OUT_BF16>
__global__ __launch_bounds__(256)
void gemm_wmma_kernel(const void* __restrict__ Aptr, const float* __restrict__ W,
                      void* __restrict__ Outptr, const float* __restrict__ bias,
                      int M, int N, int K) {
    __shared__ unsigned short sA[2][128 * 32];
    __shared__ unsigned short sB[2][128 * 32];

    const int tid  = threadIdx.x;
    const int lane = tid & 31;
    const int wave = tid >> 5;
    const int mw   = wave & 3;           // wave's 32-row group
    const int nw   = wave >> 2;          // wave's 64-col group
    const int row0 = blockIdx.x * 128;
    const int col0 = blockIdx.y * 128;
    const float* Af = (const float*)Aptr;

    Acc acc[2][4];
#pragma unroll
    for (int i = 0; i < 2; ++i)
#pragma unroll
        for (int j = 0; j < 4; ++j)
#pragma unroll
            for (int r = 0; r < 8; ++r) acc[i][j].f[r] = 0.0f;

    float areg[16], wreg[16];
    const int nt = K / 32;

    // ---- stage tile 0 ----
    if (A_BF16) {
        if (wave == 0)
            tdm_load_2d_bf16((unsigned int)(size_t)(void*)sA[0],
                             (const unsigned short*)Aptr + (size_t)row0 * K,
                             32, 128, 32, 128, (unsigned long long)K);
    } else {
#pragma unroll
        for (int j = 0; j < 4; ++j) {
            const int idx = tid + 256 * j, r = idx >> 3, c4 = idx & 7;
            const float4 v = *(const float4*)(Af + (size_t)(row0 + r) * K + c4 * 4);
            areg[4 * j] = v.x; areg[4 * j + 1] = v.y; areg[4 * j + 2] = v.z; areg[4 * j + 3] = v.w;
        }
    }
#pragma unroll
    for (int j = 0; j < 4; ++j) {
        const int idx = tid + 256 * j, r = idx >> 3, c4 = idx & 7;
        const float4 v = *(const float4*)(W + (size_t)(col0 + r) * K + c4 * 4);
        wreg[4 * j] = v.x; wreg[4 * j + 1] = v.y; wreg[4 * j + 2] = v.z; wreg[4 * j + 3] = v.w;
    }
    if (!A_BF16) {
#pragma unroll
        for (int j = 0; j < 4; ++j) {
            const int idx = tid + 256 * j, r = idx >> 3, c4 = idx & 7;
            uint2 p = { pack_bf16x2(areg[4 * j], areg[4 * j + 1]),
                        pack_bf16x2(areg[4 * j + 2], areg[4 * j + 3]) };
            *(uint2*)(&sA[0][r * 32 + c4 * 4]) = p;
        }
    }
#pragma unroll
    for (int j = 0; j < 4; ++j) {
        const int idx = tid + 256 * j, r = idx >> 3, c4 = idx & 7;
        uint2 p = { pack_bf16x2(wreg[4 * j], wreg[4 * j + 1]),
                    pack_bf16x2(wreg[4 * j + 2], wreg[4 * j + 3]) };
        *(uint2*)(&sB[0][r * 32 + c4 * 4]) = p;
    }
    if (A_BF16 && wave == 0) __builtin_amdgcn_s_wait_tensorcnt(0);
    __syncthreads();

    // ---- main loop, double buffered ----
    for (int t = 0; t < nt; ++t) {
        const int cur = t & 1, nxt = cur ^ 1;
        const bool more = (t + 1) < nt;
        const int kb = (t + 1) * 32;

        if (more) {
            if (A_BF16) {
                if (wave == 0)
                    tdm_load_2d_bf16((unsigned int)(size_t)(void*)sA[nxt],
                                     (const unsigned short*)Aptr + (size_t)row0 * K + kb,
                                     32, 128, 32, 128, (unsigned long long)K);
            } else {
#pragma unroll
                for (int j = 0; j < 4; ++j) {
                    const int idx = tid + 256 * j, r = idx >> 3, c4 = idx & 7;
                    const float4 v = *(const float4*)(Af + (size_t)(row0 + r) * K + kb + c4 * 4);
                    areg[4 * j] = v.x; areg[4 * j + 1] = v.y; areg[4 * j + 2] = v.z; areg[4 * j + 3] = v.w;
                }
            }
#pragma unroll
            for (int j = 0; j < 4; ++j) {
                const int idx = tid + 256 * j, r = idx >> 3, c4 = idx & 7;
                const float4 v = *(const float4*)(W + (size_t)(col0 + r) * K + kb + c4 * 4);
                wreg[4 * j] = v.x; wreg[4 * j + 1] = v.y; wreg[4 * j + 2] = v.z; wreg[4 * j + 3] = v.w;
            }
        }

        // compute current tile: 32x64 wave tile = 2 A-frags x 4 B-frags, 8 WMMA
        const Frag16 a0 = load_a_frag(&sA[cur][(mw * 32) * 32], 32, lane);
        const Frag16 a1 = load_a_frag(&sA[cur][(mw * 32 + 16) * 32], 32, lane);
#pragma unroll
        for (int j = 0; j < 4; ++j) {
            const Frag16 b = load_b_frag(&sB[cur][(nw * 64 + j * 16) * 32], 32, lane);
            acc[0][j].v = wmma_bf16(a0, b, acc[0][j].v);
            acc[1][j].v = wmma_bf16(a1, b, acc[1][j].v);
        }

        if (more) {
            if (!A_BF16) {
#pragma unroll
                for (int j = 0; j < 4; ++j) {
                    const int idx = tid + 256 * j, r = idx >> 3, c4 = idx & 7;
                    uint2 p = { pack_bf16x2(areg[4 * j], areg[4 * j + 1]),
                                pack_bf16x2(areg[4 * j + 2], areg[4 * j + 3]) };
                    *(uint2*)(&sA[nxt][r * 32 + c4 * 4]) = p;
                }
            }
#pragma unroll
            for (int j = 0; j < 4; ++j) {
                const int idx = tid + 256 * j, r = idx >> 3, c4 = idx & 7;
                uint2 p = { pack_bf16x2(wreg[4 * j], wreg[4 * j + 1]),
                            pack_bf16x2(wreg[4 * j + 2], wreg[4 * j + 3]) };
                *(uint2*)(&sB[nxt][r * 32 + c4 * 4]) = p;
            }
            if (A_BF16 && wave == 0) __builtin_amdgcn_s_wait_tensorcnt(0);
        }
        __syncthreads();
    }

    // ---- epilogue: C layout (VGPR r: row r+8*(lane>=16), col lane&15) ----
    const int g = lane >> 4, n = lane & 15;
#pragma unroll
    for (int i = 0; i < 2; ++i) {
#pragma unroll
        for (int j = 0; j < 4; ++j) {
#pragma unroll
            for (int r = 0; r < 8; ++r) {
                const int gm = row0 + mw * 32 + i * 16 + r + 8 * g;
                const int gn = col0 + nw * 64 + j * 16 + n;
                if (OUT_BF16)
                    ((unsigned short*)Outptr)[(size_t)gm * N + gn] = f32_to_bf16(acc[i][j].f[r]);
                else
                    ((float*)Outptr)[(size_t)gm * N + gn] = acc[i][j].f[r] + bias[gn];
            }
        }
    }
}

// ---------------------------------------------------------------------------
// Flash attention (causal). qkv: bf16 [B*T, 3*D] packed Q|K|V per row.
// Block = (b, h, 128 q rows), 8 waves x 16 q rows.  K tile (32x64) staged by
// the Tensor Data Mover; V tile manually transposed into LDS.
// ---------------------------------------------------------------------------
__global__ __launch_bounds__(256)
void flash_attn_kernel(const unsigned short* __restrict__ qkv,
                       unsigned short* __restrict__ out,
                       int Bb, int T, int H) {
    const int D = 1024, E = 3072;
    __shared__ unsigned short sK[32 * 64];       // [key][hd]   (TDM target)
    __shared__ unsigned short sVt[64 * 32];      // [hd][key]   (transposed)
    __shared__ unsigned short sP[8 * 16 * 32];   // per-wave P staging

    const int tid  = threadIdx.x;
    const int lane = tid & 31;
    const int wave = tid >> 5;
    const int b = blockIdx.z, h = blockIdx.y;
    const int q_base = blockIdx.x * 128;
    const int wq = q_base + wave * 16;
    const int m = lane & 15, g = lane >> 4, n = lane & 15;
    const float scale = 0.125f;                  // 1/sqrt(64)

    // Q fragments (16x64 = two 16x32 A-frags) straight from global bf16.
    Frag16 aq[2];
#pragma unroll
    for (int f = 0; f < 2; ++f)
#pragma unroll
        for (int v = 0; v < 8; ++v) {
            const int k0 = a_k0(v, g) + f * 32;
            const size_t off = (size_t)(b * T + wq + m) * E + h * 64 + k0;
            aq[f].u[v] = *(const unsigned int*)(qkv + off);
        }

    float ms[8], ls[8];
    Acc o[4];
#pragma unroll
    for (int r = 0; r < 8; ++r) { ms[r] = -1e30f; ls[r] = 0.0f; }
#pragma unroll
    for (int j = 0; j < 4; ++j)
#pragma unroll
        for (int r = 0; r < 8; ++r) o[j].f[r] = 0.0f;

    const int nkt = (q_base + 128) / 32;
    for (int kt = 0; kt < nkt; ++kt) {
        const int k0g = kt * 32;

        // K tile via TDM: 32 rows x 64 bf16, row stride 3072 elements.
        if (wave == 0)
            tdm_load_2d_bf16((unsigned int)(size_t)(void*)sK,
                             qkv + (size_t)(b * T + k0g) * E + 1024 + h * 64,
                             64, 32, 64, (unsigned int)T, (unsigned long long)E);

        // V tile manual transposed stage (TDM cannot transpose).
#pragma unroll
        for (int j = 0; j < 4; ++j) {
            const int idx = tid + 256 * j;       // 0..1023 -> 32 keys x 32 col pairs
            const int key = idx >> 5;
            const int c = (idx & 31) * 2;
            const size_t base = (size_t)(b * T + k0g + key) * E + h * 64 + c;
            const unsigned int vv = *(const unsigned int*)(qkv + base + 2048);
            sVt[c * 32 + key]       = (unsigned short)(vv & 0xFFFFu);
            sVt[(c + 1) * 32 + key] = (unsigned short)(vv >> 16);
        }
        if (wave == 0) __builtin_amdgcn_s_wait_tensorcnt(0);
        __syncthreads();

        if (k0g <= wq + 15) {                    // wave-uniform causal skip
            Acc s[2];
#pragma unroll
            for (int nb = 0; nb < 2; ++nb) {
#pragma unroll
                for (int r = 0; r < 8; ++r) s[nb].f[r] = 0.0f;
#pragma unroll
                for (int f = 0; f < 2; ++f) {
                    const Frag16 bk = load_b_frag(sK + (nb * 16) * 64 + f * 32, 64, lane);
                    s[nb].v = wmma_bf16(aq[f], bk, s[nb].v);
                }
            }

            // Online softmax in accumulator layout.
            float p0[8], p1[8];
#pragma unroll
            for (int r = 0; r < 8; ++r) {
                const int q_idx = wq + r + 8 * g;
                float s0 = s[0].f[r] * scale;
                float s1 = s[1].f[r] * scale;
                if (k0g + n > q_idx)      s0 = -1e30f;
                if (k0g + 16 + n > q_idx) s1 = -1e30f;
                const float rmax  = half_reduce_max(fmaxf(s0, s1));
                const float nm    = fmaxf(ms[r], rmax);
                const float alpha = __expf(ms[r] - nm);
                const float e0 = __expf(s0 - nm);
                const float e1 = __expf(s1 - nm);
                const float rsum = half_reduce_sum(e0 + e1);
                ls[r] = ls[r] * alpha + rsum;
                ms[r] = nm;
#pragma unroll
                for (int j = 0; j < 4; ++j) o[j].f[r] *= alpha;
                p0[r] = e0;
                p1[r] = e1;
            }

            // P (C-layout f32) -> bf16 LDS -> A-fragment.
            unsigned short* pp = sP + wave * 512;
#pragma unroll
            for (int r = 0; r < 8; ++r) {
                const int mm = r + 8 * g;
                pp[mm * 32 + n]      = f32_to_bf16(p0[r]);
                pp[mm * 32 + 16 + n] = f32_to_bf16(p1[r]);
            }
            __builtin_amdgcn_wave_barrier();
            asm volatile("s_wait_dscnt 0" ::: "memory");   // wave-local LDS RAW fence
            const Frag16 pf = load_a_frag(pp, 32, lane);

            // O += P V
#pragma unroll
            for (int j = 0; j < 4; ++j) {
                const Frag16 bv = load_b_frag(sVt + (j * 16) * 32, 32, lane);
                o[j].v = wmma_bf16(pf, bv, o[j].v);
            }
        }
        __syncthreads();
    }

    // O / l -> bf16 [B*T, D]
#pragma unroll
    for (int j = 0; j < 4; ++j)
#pragma unroll
        for (int r = 0; r < 8; ++r) {
            const int q_idx = wq + r + 8 * g;
            const int col = h * 64 + j * 16 + n;
            out[(size_t)(b * T + q_idx) * D + col] = f32_to_bf16(o[j].f[r] / ls[r]);
        }
}

// ---------------------------------------------------------------------------
extern "C" void kernel_launch(void* const* d_in, const int* in_sizes, int n_in,
                              void* d_out, int out_size, void* d_ws, size_t ws_size,
                              hipStream_t stream) {
    (void)in_sizes; (void)n_in; (void)out_size; (void)ws_size;

    const float* x      = (const float*)d_in[0];   // [B,T,D] fp32
    const float* w_qkv  = (const float*)d_in[1];   // [3D,D] fp32
    const float* w_proj = (const float*)d_in[2];   // [D,D]  fp32
    const float* b_proj = (const float*)d_in[3];   // [D]    fp32
    float* out = (float*)d_out;                    // [B,T,D] fp32

    const int Bb = 2, T = 2048, D = 1024, H = 16;
    const int M = Bb * T;                          // 4096

    unsigned short* qkv_ws  = (unsigned short*)d_ws;                       // bf16 [M, 3D]
    unsigned short* attn_ws = (unsigned short*)((char*)d_ws +
                              (size_t)M * 3 * D * sizeof(unsigned short)); // bf16 [M, D]

    // 1) QKV projection: [M,D] x [3D,D]^T -> bf16 [M,3D]
    gemm_wmma_kernel<false, true><<<dim3(M / 128, (3 * D) / 128), 256, 0, stream>>>(
        x, w_qkv, qkv_ws, nullptr, M, 3 * D, D);

    // 2) Causal flash attention -> bf16 [M,D] (already [B,T,H*hd] layout)
    flash_attn_kernel<<<dim3(T / 128, H, Bb), 256, 0, stream>>>(qkv_ws, attn_ws, Bb, T, H);

    // 3) Output projection + bias (A staged via TDM): [M,D] x [D,D]^T -> fp32 [M,D]
    gemm_wmma_kernel<true, false><<<dim3(M / 128, D / 128), 256, 0, stream>>>(
        attn_ws, w_proj, out, b_proj, M, D, D);
}
